// GatedCrossAttentionBlock_52415780880961
// MI455X (gfx1250) — matrix operand order repaired and checked by reading, not compiled, not run on hardware
//
#include <hip/hip_runtime.h>
#include <hip/hip_bf16.h>

// ---------------- problem constants (from reference) ----------------
#define BS    2
#define EMB   256
#define SEQ   1024
#define TTOT  8525          // 80*80+40*40+20*20+10*10+5*5
#define NH    8
#define HD    32            // EMB / NH

typedef __bf16 bf16_t;
typedef bf16_t v16bf __attribute__((ext_vector_type(16)));
typedef bf16_t v8bf  __attribute__((ext_vector_type(8)));
typedef float  v8f   __attribute__((ext_vector_type(8)));

__device__ __forceinline__ bf16_t f2bf(float f) {
  unsigned u = __float_as_uint(f);
  u += 0x7FFFu + ((u >> 16) & 1u);           // round-to-nearest-even
  unsigned short s = (unsigned short)(u >> 16);
  return __builtin_bit_cast(bf16_t, s);
}

__device__ __forceinline__ v16bf cat8(v8bf lo, v8bf hi) {
  return __builtin_shufflevector(lo, hi, 0,1,2,3,4,5,6,7,8,9,10,11,12,13,14,15);
}

__device__ __forceinline__ float wave_sum32(float x) {
  #pragma unroll
  for (int m = 16; m >= 1; m >>= 1) x += __shfl_xor(x, m, 32);
  return x;
}

__device__ __forceinline__ void level_of(int t, int& start, int& hw) {
  if      (t < 6400) { start = 0;    hw = 6400; }
  else if (t < 8000) { start = 6400; hw = 1600; }
  else if (t < 8400) { start = 8000; hw = 400;  }
  else if (t < 8500) { start = 8400; hw = 100;  }
  else               { start = 8500; hw = 25;   }
}

// ---------- kernel 0: one-time f32 -> bf16 conversion (weights) ----------
__global__ void __launch_bounds__(256)
cvt_bf16_kernel(const float* __restrict__ in, bf16_t* __restrict__ out, int n) {
  int i = blockIdx.x * 256 + threadIdx.x;
  if (i < n) out[i] = f2bf(in[i]);
}

// ---------- kernel 1: gather feature maps into v rows + LayerNorm ----------
// one wave per (b,t) row; lane covers 8 channels (ch = lane + 32j)
__global__ void __launch_bounds__(256)
build_v_ln_kernel(const float* __restrict__ q0, const float* __restrict__ q1,
                  const float* __restrict__ q2, const float* __restrict__ q3,
                  const float* __restrict__ q4,
                  const float* __restrict__ g, const float* __restrict__ bta,
                  float* __restrict__ v, bf16_t* __restrict__ v_bf,
                  bf16_t* __restrict__ vn_bf) {
  int row = blockIdx.x * 8 + (threadIdx.x >> 5);
  if (row >= BS * TTOT) return;
  int lane = threadIdx.x & 31;
  int b = row / TTOT, t = row % TTOT;
  int start, hw; level_of(t, start, hw);
  const float* src = (start == 0) ? q0 : (start == 6400) ? q1 :
                     (start == 8000) ? q2 : (start == 8400) ? q3 : q4;
  int p = t - start;
  float x[8]; float s = 0.f;
  #pragma unroll
  for (int j = 0; j < 8; ++j) {
    int ch = lane + 32 * j;
    x[j] = src[((size_t)(b * EMB + ch)) * hw + p];
    s += x[j];
  }
  float mean = wave_sum32(s) * (1.0f / EMB);
  float vs = 0.f;
  #pragma unroll
  for (int j = 0; j < 8; ++j) { float d = x[j] - mean; vs += d * d; }
  float inv = rsqrtf(wave_sum32(vs) * (1.0f / EMB) + 1e-5f);
  size_t base = (size_t)row * EMB;
  #pragma unroll
  for (int j = 0; j < 8; ++j) {
    int ch = lane + 32 * j;
    v    [base + ch] = x[j];
    v_bf [base + ch] = f2bf(x[j]);
    vn_bf[base + ch] = f2bf((x[j] - mean) * inv * g[ch] + bta[ch]);
  }
}

// ---------- kernel 2: row LayerNorm for the cache -> bf16 ----------
__global__ void __launch_bounds__(256)
ln_rows_kernel(const float* __restrict__ in, const float* __restrict__ g,
               const float* __restrict__ bta, bf16_t* __restrict__ outp, int rows) {
  int row = blockIdx.x * 8 + (threadIdx.x >> 5);
  if (row >= rows) return;
  int lane = threadIdx.x & 31;
  size_t base = (size_t)row * EMB;
  float x[8]; float s = 0.f;
  #pragma unroll
  for (int j = 0; j < 8; ++j) { x[j] = in[base + lane + 32 * j]; s += x[j]; }
  float mean = wave_sum32(s) * (1.0f / EMB);
  float vs = 0.f;
  #pragma unroll
  for (int j = 0; j < 8; ++j) { float d = x[j] - mean; vs += d * d; }
  float inv = rsqrtf(wave_sum32(vs) * (1.0f / EMB) + 1e-5f);
  #pragma unroll
  for (int j = 0; j < 8; ++j) {
    int ch = lane + 32 * j;
    outp[base + ch] = f2bf((x[j] - mean) * inv * g[ch] + bta[ch]);
  }
}

// ---------- kernel 3: WMMA-bf16 GEMM, 16x64 tile per wave ----------
// A: M x K bf16 row-major, B: K x N bf16 row-major. 4 accumulators share one
// A fragment per K-step (loads/wmma = 1.5 instead of 4).
// mode: 0 f32 out, 1 bf16 out *scale, 2 bf16 out, 3 bf16 relu,
//       4 f32 tanh, 5 bf16 per-head-transposed store (K cache: [b,h,d,s])
__global__ void __launch_bounds__(256)
gemm_kernel(const bf16_t* __restrict__ A, const bf16_t* __restrict__ B,
            const float* __restrict__ bias, void* __restrict__ Cout,
            int M, int N, int K, int mode, float scale) {
  int wv = threadIdx.x >> 5, lane = threadIdx.x & 31;
  int ng = N >> 6;                                   // 64-col groups
  int mt = (M + 15) >> 4;
  int tile = blockIdx.x * 8 + wv;
  if (tile >= mt * ng) return;
  int tm = tile / ng, tg = tile % ng;
  int off = (lane >> 4) * 8;                         // K-half select for A layout
  int ma = tm * 16 + (lane & 15); if (ma >= M) ma = M - 1;  // clamp for loads
  const bf16_t* Arow = A + (size_t)ma * K;
  const bf16_t* Bcol = B + tg * 64;
  v8f acc0 = {}, acc1 = {}, acc2 = {}, acc3 = {};
  for (int kb = 0; kb < K; kb += 32) {
    // A fragment: rows m=lane&15, k = (e&7)+(e>=8?16:0)+off -> two 16B runs
    v16bf av = cat8(*(const v8bf*)(Arow + kb + off),
                    *(const v8bf*)(Arow + kb + off + 16));
    // B fragments: col n = lane&15 per 16-col group, k = (e&7)+(e>=8?16:0)+off
    const bf16_t* Bc = Bcol + (lane & 15);
    v16bf b0, b1, b2, b3;
    #pragma unroll
    for (int e = 0; e < 16; ++e) {
      const bf16_t* Brow = Bc + (size_t)(kb + (e & 7) + (e >= 8 ? 16 : 0) + off) * N;
      b0[e] = Brow[0]; b1[e] = Brow[16]; b2[e] = Brow[32]; b3[e] = Brow[48];
    }
    acc0 = __builtin_amdgcn_wmma_f32_16x16x32_bf16(false, av, false, b0, (short)0, acc0, false, false);
    acc1 = __builtin_amdgcn_wmma_f32_16x16x32_bf16(false, av, false, b1, (short)0, acc1, false, false);
    acc2 = __builtin_amdgcn_wmma_f32_16x16x32_bf16(false, av, false, b2, (short)0, acc2, false, false);
    acc3 = __builtin_amdgcn_wmma_f32_16x16x32_bf16(false, av, false, b3, (short)0, acc3, false, false);
  }
  v8f accs[4] = {acc0, acc1, acc2, acc3};
  #pragma unroll
  for (int j = 0; j < 4; ++j) {
    int n = tg * 64 + j * 16 + (lane & 15);
    float bb = bias ? bias[n] : 0.0f;
    #pragma unroll
    for (int r = 0; r < 8; ++r) {
      int m = tm * 16 + r + 8 * (lane >> 4);
      if (m >= M) continue;
      float x = accs[j][r] + bb;
      if      (mode == 1) x *= scale;
      else if (mode == 3) x = fmaxf(x, 0.0f);
      else if (mode == 4) x = tanhf(x);
      if (mode == 0 || mode == 4) {
        ((float*)Cout)[(size_t)m * N + n] = x;
      } else if (mode == 5) {          // m = b*SEQ + s, n = h*HD + d -> [b,h,d,s]
        int b = m / SEQ, s = m % SEQ;
        int h = n / HD,  d = n % HD;
        ((bf16_t*)Cout)[((size_t)(b * NH + h) * HD + d) * SEQ + s] = f2bf(x);
      } else {
        ((bf16_t*)Cout)[(size_t)m * N + n] = f2bf(x);
      }
    }
  }
}

// ---------- kernel 4: flash attention (streaming softmax, all bf16 operands) ----------
// One wave per (b, h, 16-row t-tile). 32 chunks of 32 keys:
// 2x QK wmma -> clip + mask bias -> online softmax -> LDS P transpose -> 2x PV wmma.
__global__ void __launch_bounds__(256)
attn_kernel(const bf16_t* __restrict__ q, const bf16_t* __restrict__ kT,
            const bf16_t* __restrict__ vmat, const int* __restrict__ mask,
            bf16_t* __restrict__ o) {
  __shared__ __align__(16) bf16_t P[8][16 * 40];   // per-wave P tile, row stride 40
  const int TT = (TTOT + 15) / 16;                 // 533
  int wv = threadIdx.x >> 5, lane = threadIdx.x & 31;
  int task = blockIdx.x * 8 + wv;
  int total = BS * NH * TT;
  bool active = task < total;
  if (!active) task = 0;                           // keep __syncthreads uniform
  int b = task / (NH * TT);
  int rem = task % (NH * TT);
  int h = rem / TT, tt = rem % TT;

  int off = (lane >> 4) * 8;
  int mrow = lane & 15;

  // Q tile in A layout (q pre-scaled by hd^-0.5); two aligned 16B loads
  int tq = tt * 16 + mrow; if (tq >= TTOT) tq = TTOT - 1;
  const bf16_t* qrow = q + ((size_t)(b * TTOT + tq)) * EMB + h * HD;
  v16bf aq = cat8(*(const v8bf*)(qrow + off), *(const v8bf*)(qrow + off + 16));

  float rmax[8], rsum[8];
  v8f accL = {}, accH = {};
  #pragma unroll
  for (int r = 0; r < 8; ++r) { rmax[r] = -3.0e38f; rsum[r] = 0.f; }

  const bf16_t* kTh = kT + (size_t)(b * NH + h) * HD * SEQ;      // row d, stride SEQ
  const bf16_t* vbb = vmat + (size_t)b * SEQ * EMB + h * HD;
  const int*    mk  = mask + b * SEQ;

  for (int sc = 0; sc < SEQ; sc += 32) {
    // QK B tiles: n = s column (lane&15), k = d = (e&7)+(e>=8?16:0)+off
    const bf16_t* krow = kTh + sc + (lane & 15);
    v16bf bk0, bk1;
    #pragma unroll
    for (int e = 0; e < 16; ++e) {
      size_t dd = (size_t)((e & 7) + (e >= 8 ? 16 : 0) + off) * SEQ;
      bk0[e] = krow[dd]; bk1[e] = krow[dd + 16];
    }
    if (sc + 32 < SEQ) __builtin_prefetch(krow + 32, 0, 0);
    v8f z = {};
    v8f c0 = __builtin_amdgcn_wmma_f32_16x16x32_bf16(false, aq, false, bk0, (short)0, z, false, false);
    v8f c1 = __builtin_amdgcn_wmma_f32_16x16x32_bf16(false, aq, false, bk1, (short)0, z, false, false);

    int s0 = sc + (lane & 15), s1 = s0 + 16;
    float mb0 = (mk[s0] == 0) ? -9.0e15f : 1.0f;   // reference adds +1.0 where mask==1
    float mb1 = (mk[s1] == 0) ? -9.0e15f : 1.0f;

    #pragma unroll
    for (int r = 0; r < 8; ++r) {
      float x0 = fminf(fmaxf(c0[r], -50000.f), 50000.f) + mb0;
      float x1 = fminf(fmaxf(c1[r], -50000.f), 50000.f) + mb1;
      float mx = fmaxf(x0, x1);
      #pragma unroll
      for (int mm = 8; mm >= 1; mm >>= 1) mx = fmaxf(mx, __shfl_xor(mx, mm, 32));
      float mnew = fmaxf(rmax[r], mx);
      float corr = __expf(rmax[r] - mnew);         // 0 on first chunk (underflow)
      float p0 = __expf(x0 - mnew);
      float p1 = __expf(x1 - mnew);
      float ps = p0 + p1;
      #pragma unroll
      for (int mm = 8; mm >= 1; mm >>= 1) ps += __shfl_xor(ps, mm, 32);
      rsum[r] = rsum[r] * corr + ps;
      rmax[r] = mnew;
      accL[r] *= corr; accH[r] *= corr;
      int m = r + 8 * (lane >> 4);                 // C-layout row of this element
      P[wv][m * 40 + (lane & 15)]      = f2bf(p0);
      P[wv][m * 40 + 16 + (lane & 15)] = f2bf(p1);
    }
    __syncthreads();
    // P tile in A layout: two aligned 16B DS loads (k runs [off,off+7], [16+off,23+off])
    v16bf ap = cat8(*(const v8bf*)&P[wv][mrow * 40 + off],
                    *(const v8bf*)&P[wv][mrow * 40 + 16 + off]);
    __syncthreads();
    // PV B tiles: n = d column (lane&15), k = s = (e&7)+(e>=8?16:0)+off
    const bf16_t* vrow = vbb + (size_t)sc * EMB + (lane & 15);
    v16bf bvL, bvH;
    #pragma unroll
    for (int e = 0; e < 16; ++e) {
      size_t ss = (size_t)((e & 7) + (e >= 8 ? 16 : 0) + off) * EMB;
      bvL[e] = vrow[ss]; bvH[e] = vrow[ss + 16];
    }
    if (sc + 32 < SEQ) __builtin_prefetch(vrow + 32 * EMB, 0, 0);
    accL = __builtin_amdgcn_wmma_f32_16x16x32_bf16(false, ap, false, bvL, (short)0, accL, false, false);
    accH = __builtin_amdgcn_wmma_f32_16x16x32_bf16(false, ap, false, bvH, (short)0, accH, false, false);
  }

  int n = lane & 15;
  #pragma unroll
  for (int r = 0; r < 8; ++r) {
    int m = r + 8 * (lane >> 4);
    int t = tt * 16 + m;
    if (active && t < TTOT) {
      float invl = 1.0f / rsum[r];
      size_t baseo = ((size_t)(b * TTOT + t)) * EMB + h * HD;
      o[baseo + n]      = f2bf(accL[r] * invl);
      o[baseo + 16 + n] = f2bf(accH[r] * invl);
    }
  }
}

// ---------- kernel 5: residual + gate, scatter to 5 output maps ----------
__global__ void __launch_bounds__(256)
final_kernel(const float* __restrict__ v, const float* __restrict__ gate,
             const float* __restrict__ delta, float* __restrict__ out) {
  size_t idx = (size_t)blockIdx.x * 256 + threadIdx.x;
  if (idx >= (size_t)BS * TTOT * EMB) return;
  int ch = (int)(idx % EMB);
  int bt = (int)(idx / EMB);
  int b = bt / TTOT, t = bt % TTOT;
  float nv = v[idx] + gate[idx] * delta[idx];
  int start, hw; level_of(t, start, hw);
  int p = t - start;
  size_t base = (size_t)BS * EMB * start;          // outputs concatenated per level
  out[base + ((size_t)(b * EMB + ch)) * hw + p] = nv;
}

// ------------------------------ launch ------------------------------
extern "C" void kernel_launch(void* const* d_in, const int* in_sizes, int n_in,
                              void* d_out, int out_size, void* d_ws, size_t ws_size,
                              hipStream_t stream) {
  const float* q0   = (const float*)d_in[0];
  const float* q1   = (const float*)d_in[1];
  const float* q2   = (const float*)d_in[2];
  const float* q3   = (const float*)d_in[3];
  const float* q4   = (const float*)d_in[4];
  const float* cache= (const float*)d_in[5];
  const int*   mask = (const int*)  d_in[6];
  const float* lnvg = (const float*)d_in[7];
  const float* lnvb = (const float*)d_in[8];
  const float* lncg = (const float*)d_in[9];
  const float* lncb = (const float*)d_in[10];
  const float* Wv   = (const float*)d_in[11];
  const float* bv   = (const float*)d_in[12];
  const float* Wc   = (const float*)d_in[13];
  const float* bc   = (const float*)d_in[14];
  const float* Wvc  = (const float*)d_in[15];
  const float* bvc  = (const float*)d_in[16];
  const float* Wo   = (const float*)d_in[17];
  const float* bo   = (const float*)d_in[18];
  const float* Wg1  = (const float*)d_in[19];
  const float* bg1  = (const float*)d_in[20];
  const float* Wg2  = (const float*)d_in[21];
  const float* bg2  = (const float*)d_in[22];

  const size_t bsTE = (size_t)BS * TTOT * EMB;     // 4,364,800
  const size_t bsSE = (size_t)BS * SEQ  * EMB;     //   524,288
  char* base = (char*)d_ws;
  size_t cur = 0;
  auto carve = [&](size_t bytes) -> char* {
    char* p = base + cur;
    cur = (cur + bytes + 255) & ~(size_t)255;
    return p;
  };
  float*  v      = (float*) carve(bsTE * 4);
  float*  gate   = (float*) carve(bsTE * 4);
  float*  delta  = (float*) carve(bsTE * 4);
  bf16_t* vn_bf  = (bf16_t*)carve(bsTE * 2);
  bf16_t* v_bf   = (bf16_t*)carve(bsTE * 2);
  bf16_t* q_bf   = (bf16_t*)carve(bsTE * 2);
  bf16_t* o_bf   = (bf16_t*)carve(bsTE * 2);
  bf16_t* g1_bf  = (bf16_t*)carve((size_t)BS * TTOT * 128 * 2);
  bf16_t* cn_bf  = (bf16_t*)carve(bsSE * 2);
  bf16_t* kT_bf  = (bf16_t*)carve(bsSE * 2);
  bf16_t* val_bf = (bf16_t*)carve(bsSE * 2);
  bf16_t* Wv_bf  = (bf16_t*)carve(EMB * EMB * 2);
  bf16_t* Wc_bf  = (bf16_t*)carve(EMB * EMB * 2);
  bf16_t* Wvc_bf = (bf16_t*)carve(EMB * EMB * 2);
  bf16_t* Wo_bf  = (bf16_t*)carve(EMB * EMB * 2);
  bf16_t* Wg1_bf = (bf16_t*)carve(EMB * 128 * 2);
  bf16_t* Wg2_bf = (bf16_t*)carve(128 * EMB * 2);

  // one-time weight conversions
  cvt_bf16_kernel<<<(EMB*EMB + 255)/256, 256, 0, stream>>>(Wv,  Wv_bf,  EMB*EMB);
  cvt_bf16_kernel<<<(EMB*EMB + 255)/256, 256, 0, stream>>>(Wc,  Wc_bf,  EMB*EMB);
  cvt_bf16_kernel<<<(EMB*EMB + 255)/256, 256, 0, stream>>>(Wvc, Wvc_bf, EMB*EMB);
  cvt_bf16_kernel<<<(EMB*EMB + 255)/256, 256, 0, stream>>>(Wo,  Wo_bf,  EMB*EMB);
  cvt_bf16_kernel<<<(EMB*128 + 255)/256, 256, 0, stream>>>(Wg1, Wg1_bf, EMB*128);
  cvt_bf16_kernel<<<(128*EMB + 255)/256, 256, 0, stream>>>(Wg2, Wg2_bf, 128*EMB);

  const int M = BS * TTOT;                         // 17050
  auto gemm_blocks = [](int m, int n) { int t = ((m + 15) / 16) * (n / 64); return (t + 7) / 8; };

  build_v_ln_kernel<<<(M + 7) / 8, 256, 0, stream>>>(q0, q1, q2, q3, q4, lnvg, lnvb,
                                                     v, v_bf, vn_bf);
  ln_rows_kernel<<<(BS * SEQ + 7) / 8, 256, 0, stream>>>(cache, lncg, lncb, cn_bf, BS * SEQ);

  // projections: q (scaled by hd^-0.5) -> bf16, k -> per-head-transposed bf16, val -> bf16
  gemm_kernel<<<gemm_blocks(M, EMB), 256, 0, stream>>>(vn_bf, Wv_bf, bv, q_bf,
                                                       M, EMB, EMB, 1, 0.17677669529663687f);
  gemm_kernel<<<gemm_blocks(BS*SEQ, EMB), 256, 0, stream>>>(cn_bf, Wc_bf,  bc,  kT_bf,
                                                            BS*SEQ, EMB, EMB, 5, 1.f);
  gemm_kernel<<<gemm_blocks(BS*SEQ, EMB), 256, 0, stream>>>(cn_bf, Wvc_bf, bvc, val_bf,
                                                            BS*SEQ, EMB, EMB, 2, 1.f);

  // flash attention: BS*NH*ceil(T/16) = 8528 wave tasks, 8 waves/block
  int tasks = BS * NH * ((TTOT + 15) / 16);
  attn_kernel<<<(tasks + 7) / 8, 256, 0, stream>>>(q_bf, kT_bf, val_bf, mask, o_bf);

  // gate path and output projection
  gemm_kernel<<<gemm_blocks(M, 128), 256, 0, stream>>>(v_bf,  Wg1_bf, bg1, g1_bf,
                                                       M, 128, EMB, 3, 1.f);
  gemm_kernel<<<gemm_blocks(M, EMB), 256, 0, stream>>>(g1_bf, Wg2_bf, bg2, gate,
                                                       M, EMB, 128, 4, 1.f);
  gemm_kernel<<<gemm_blocks(M, EMB), 256, 0, stream>>>(o_bf,  Wo_bf,  bo,  delta,
                                                       M, EMB, EMB, 0, 1.f);

  final_kernel<<<(int)((bsTE + 255) / 256), 256, 0, stream>>>(v, gate, delta, (float*)d_out);
}
